// DetailDecoder_45440753992333
// MI455X (gfx1250) — matrix operand order repaired and checked by reading, not compiled, not run on hardware
//
#include <hip/hip_runtime.h>
#include <math.h>

// ---------------------------------------------------------------------------
// MI455X (gfx1250) transformer forward, v3.1:
//  - all GEMM/attention matmuls on v_wmma_f32_16x16x32_f16 (f32 accumulate)
//  - weights converted once per launch to f16 transposed [N,K]
//  - activations flow as f16 between matmuls; residual stream stays fp32
//  - LDS staging is pure 16B copies via GLOBAL_LOAD_ASYNC_TO_LDS_B128
//    (typed-pointer builtin signature), sync h8-copy fallback
//  - double-buffered tiles, one barrier per K-step
// ---------------------------------------------------------------------------

typedef __attribute__((ext_vector_type(16))) _Float16 v16h;
typedef __attribute__((ext_vector_type(8)))  float    v8f;
typedef __attribute__((ext_vector_type(4)))  float    f4;
typedef __attribute__((ext_vector_type(8)))  _Float16 h8;
typedef __attribute__((ext_vector_type(4)))  _Float16 h4;
typedef int v4i_gcc __attribute__((vector_size(16)));

#if defined(__has_builtin)
#if __has_builtin(__builtin_amdgcn_global_load_async_to_lds_b128) && \
    __has_builtin(__builtin_amdgcn_s_wait_asynccnt)
#define USE_ASYNC 1
#endif
#endif
#ifndef USE_ASYNC
#define USE_ASYNC 0
#endif

__device__ inline void async_ld16(const _Float16* g, _Float16* l) {
#if USE_ASYNC
  __builtin_amdgcn_global_load_async_to_lds_b128(
      (v4i_gcc __attribute__((address_space(1)))*)g,
      (v4i_gcc __attribute__((address_space(3)))*)l, 0, 0);
#else
  *(h8*)l = *(const h8*)g;
#endif
}
__device__ inline void async_wait() {
#if USE_ASYNC
  __builtin_amdgcn_s_wait_asynccnt(0);
#endif
}

__device__ inline v8f wmma16(v16h a, v16h b, v8f c) {
  return __builtin_amdgcn_wmma_f32_16x16x32_f16(
      false, a, false, b, (short)0, c, false, false);
}
__device__ inline v16h cat16(h8 a, h8 b) {
  return __builtin_shufflevector(a, b, 0, 1, 2, 3, 4, 5, 6, 7,
                                 8, 9, 10, 11, 12, 13, 14, 15);
}

constexpr int kB   = 8;
constexpr int kT   = 1024;
constexpr int kDM  = 512;
constexpr int kNH  = 8;
constexpr int kDH  = 64;
constexpr int kDFF = 2048;
constexpr int kNL  = 8;
constexpr int kBT  = kB * kT;
constexpr int kDIN = 141;
constexpr int kKIN = 160;
constexpr int kDOUT = 139;

// ---------------------------------------------------------------------------
// Weight convert fp32 [L,Kb,N] -> f16 transposed [L,N,Kpad] (zero-padded K)
// ---------------------------------------------------------------------------
__global__ void k_cvt_w(const float* __restrict__ W, _Float16* __restrict__ Wt,
                        int Kb, int N, int Kpad, long tot) {
  long i = (long)blockIdx.x * blockDim.x + threadIdx.x;
  if (i >= tot) return;
  int kk = (int)(i % Kpad);
  long nn = i / Kpad;
  int n = (int)(nn % N);
  int l = (int)(nn / N);
  float v = (kk < Kb) ? W[((size_t)l * Kb + kk) * N + n] : 0.f;
  Wt[i] = (_Float16)v;
}

// ---------------------------------------------------------------------------
// Build padded f16 input x0[BT,160] = concat(motion, traj, mask, zeros)
// ---------------------------------------------------------------------------
__global__ void k_build(const float* __restrict__ motion,
                        const float* __restrict__ traj,
                        const float* __restrict__ mask,
                        _Float16* __restrict__ x0) {
  int i = blockIdx.x * blockDim.x + threadIdx.x;
  if (i >= kBT * kKIN) return;
  int row = i / kKIN, c = i % kKIN;
  float v;
  if (c < 135)      v = motion[(size_t)row * 135 + c];
  else if (c < 140) v = traj[(size_t)row * 5 + (c - 135)];
  else if (c == 140) v = mask[row];
  else v = 0.f;
  x0[i] = (_Float16)v;
}

// ---------------------------------------------------------------------------
// WMMA GEMM: A[M,K] f16  x  Bt[N,K] f16 (transposed weights) + bias
// mode 0: fp32 C (+opt residual) ; mode 1: f16 Ch row-major ;
// mode 2: f16 Ch in attention-V layout [b,h,d,t] (packed 16B stores)
// act: 0 none, 1 prelu(alpha), 2 relu
// ---------------------------------------------------------------------------
__global__ __launch_bounds__(128)
void k_gemm(const _Float16* __restrict__ A, const _Float16* __restrict__ Bt,
            const float* __restrict__ bias, const float* __restrict__ res,
            float* __restrict__ C, _Float16* __restrict__ Ch,
            int N, int K, int mode, int act, const float* __restrict__ alphap) {
  __shared__ alignas(16) _Float16 Ah[2][64 * 32];   // [row][k]
  __shared__ alignas(16) _Float16 Bl[2][64 * 32];   // [col][k]
  const int tid = threadIdx.x, lane = tid & 31, wid = tid >> 5;
  const int m0 = blockIdx.x * 64, n0 = blockIdx.y * 64;
  const float alpha = (act == 1) ? alphap[0] : 0.f;
  const bool async_ok = USE_ASYNC && (n0 + 64 <= N);

  auto stage = [&](int k0, int buf) {
    if (async_ok) {
#pragma unroll
      for (int i = 0; i < 2; ++i) {
        int e = tid + i * 128;                 // 256 chunks of 8 halfs each
        int r = e >> 2, c8 = (e & 3) * 8;
        async_ld16(A  + (size_t)(m0 + r) * K + k0 + c8, &Ah[buf][r * 32 + c8]);
        async_ld16(Bt + (size_t)(n0 + r) * K + k0 + c8, &Bl[buf][r * 32 + c8]);
      }
    } else {
      h8 ta[2], tb[2];
#pragma unroll
      for (int i = 0; i < 2; ++i) {
        int e = tid + i * 128;
        int r = e >> 2, c8 = (e & 3) * 8;
        ta[i] = *(const h8*)(A + (size_t)(m0 + r) * K + k0 + c8);
        tb[i] = ((n0 + r) < N) ? *(const h8*)(Bt + (size_t)(n0 + r) * K + k0 + c8)
                               : h8{};
      }
#pragma unroll
      for (int i = 0; i < 2; ++i) {
        int e = tid + i * 128;
        int r = e >> 2, c8 = (e & 3) * 8;
        *(h8*)(&Ah[buf][r * 32 + c8]) = ta[i];
        *(h8*)(&Bl[buf][r * 32 + c8]) = tb[i];
      }
    }
  };

  v8f acc[4] = {v8f{}, v8f{}, v8f{}, v8f{}};
  const int nk = K >> 5;
  stage(0, 0);

  for (int it = 0; it < nk; ++it) {
    const int buf = it & 1;
    async_wait();
    __syncthreads();
    if (it + 1 < nk) stage((it + 1) * 32, buf ^ 1);

    const int arow = wid * 16 + (lane & 15);
    const int kofs = (lane >> 4) * 8;
    const int kb   = (lane >> 4) * 16;
    v16h af = cat16(*(const h8*)&Ah[buf][arow * 32 + kofs],
                    *(const h8*)&Ah[buf][arow * 32 + kofs + 16]);
    v16h bf[4];
#pragma unroll
    for (int n = 0; n < 4; ++n) {
      int col = n * 16 + (lane & 15);
      bf[n] = cat16(*(const h8*)&Bl[buf][col * 32 + kofs],
                    *(const h8*)&Bl[buf][col * 32 + kofs + 16]);
    }
#pragma unroll
    for (int n = 0; n < 4; ++n) acc[n] = wmma16(af, bf[n], acc[n]);
  }

  // epilogue: VGPR r, lane -> row rbase+r, col lane&15
  const int rbase = m0 + wid * 16 + 8 * (lane >> 4);
#pragma unroll
  for (int n = 0; n < 4; ++n) {
    int col = n0 + n * 16 + (lane & 15);
    if (col >= N) continue;
    float bv = bias ? bias[col] : 0.f;
    float vr[8];
#pragma unroll
    for (int r = 0; r < 8; ++r) {
      float v = acc[n][r] + bv;
      if (act == 1)      v = (v > 0.f) ? v : alpha * v;
      else if (act == 2) v = (v > 0.f) ? v : 0.f;
      vr[r] = v;
    }
    if (mode == 0) {
#pragma unroll
      for (int r = 0; r < 8; ++r) {
        int row = rbase + r;
        float v = vr[r];
        if (res) v += res[(size_t)row * N + col];
        C[(size_t)row * N + col] = v;
      }
    } else if (mode == 1) {
#pragma unroll
      for (int r = 0; r < 8; ++r)
        Ch[(size_t)(rbase + r) * N + col] = (_Float16)vr[r];
    } else {
      // V layout: [b, h, d, t]; 8 consecutive t per lane -> one 16B store
      int hh = col >> 6, dd = col & 63;
      int bbq = rbase >> 10, t0 = rbase & 1023;
      h8 pk;
#pragma unroll
      for (int r = 0; r < 8; ++r) pk[r] = (_Float16)vr[r];
      *(h8*)(Ch + ((size_t)(bbq * kNH + hh) * kDH + dd) * kT + t0) = pk;
    }
  }
}

// ---------------------------------------------------------------------------
// Periodic positional encoding added in place to fp32 x[BT,512]
// ---------------------------------------------------------------------------
__global__ void k_pe(float* __restrict__ x) {
  int i = blockIdx.x * blockDim.x + threadIdx.x;
  if (i >= kBT * kDM) return;
  int c = i & (kDM - 1);
  int t = (i / kDM) % kT;
  float pos = (float)(t % 30);
  float div = __expf(-logf(10000.f) * (float)(c & ~1) / (float)kDM);
  float ang = pos * div;
  x[i] += (c & 1) ? __cosf(ang) : __sinf(ang);
}

// ---------------------------------------------------------------------------
// LayerNorm: wave per row; fp32 in, f16 out (feeds GEMMs)
// ---------------------------------------------------------------------------
__global__ __launch_bounds__(128)
void k_ln(const float* __restrict__ x, const float* __restrict__ g,
          const float* __restrict__ bt, _Float16* __restrict__ y) {
  int row  = blockIdx.x * 4 + (threadIdx.x >> 5);
  int lane = threadIdx.x & 31;
  const f4* xr = (const f4*)(x + (size_t)row * kDM);
  f4 vals[4];
  float s = 0.f;
#pragma unroll
  for (int i = 0; i < 4; ++i) {
    vals[i] = xr[lane + i * 32];
    s += vals[i][0] + vals[i][1] + vals[i][2] + vals[i][3];
  }
#pragma unroll
  for (int off = 16; off >= 1; off >>= 1) s += __shfl_xor(s, off, 32);
  float mean = s * (1.f / kDM);
  float vs = 0.f;
#pragma unroll
  for (int i = 0; i < 4; ++i)
#pragma unroll
    for (int j = 0; j < 4; ++j) { float d = vals[i][j] - mean; vs += d * d; }
#pragma unroll
  for (int off = 16; off >= 1; off >>= 1) vs += __shfl_xor(vs, off, 32);
  float inv = rsqrtf(vs * (1.f / kDM) + 1e-5f);
  const f4* g4 = (const f4*)g;
  const f4* b4 = (const f4*)bt;
#pragma unroll
  for (int i = 0; i < 4; ++i) {
    int c = lane + i * 32;
    f4 gg = g4[c], bb = b4[c];
    h4 o;
#pragma unroll
    for (int j = 0; j < 4; ++j)
      o[j] = (_Float16)(gg[j] * (vals[i][j] - mean) * inv + bb[j]);
    *(h4*)(y + (size_t)row * kDM + c * 4) = o;
  }
}

// ---------------------------------------------------------------------------
// Flash attention. q,k: f16 [B*T, 512]; v: f16 [b,h,d,t]; out: f16 [B*T,512].
// grid = (T/64, B*H); 4 waves; double-buffered pure-copy K/V tiles.
// ---------------------------------------------------------------------------
__global__ __launch_bounds__(128)
void k_attn(const _Float16* __restrict__ q, const _Float16* __restrict__ k,
            const _Float16* __restrict__ vt, _Float16* __restrict__ o) {
  __shared__ alignas(16) _Float16 Ks[2][32 * 64];   // [key][d]
  __shared__ alignas(16) _Float16 Vt[2][64 * 32];   // [d][key]
  __shared__ alignas(16) _Float16 Pw[4][16 * 32];   // per-wave probs

  const int tid = threadIdx.x, lane = tid & 31, wid = tid >> 5;
  const int bh = blockIdx.y, bb = bh >> 3, h = bh & 7;
  const float bias_s = -exp2f(-(float)(h + 1));
  const size_t base = (size_t)bb * kT * kDM + (size_t)h * kDH;
  const size_t vbase = (size_t)bh * kDH * kT;
  const int q0 = blockIdx.x * 64;
  const int qrow_w = q0 + wid * 16;
  const int g = lane >> 4;

  auto stageKV = [&](int kb, int buf) {
#if USE_ASYNC
#pragma unroll
    for (int i = 0; i < 2; ++i) {
      int e = tid + i * 128;
      int key = e >> 3, c8 = (e & 7) * 8;         // Ks: 32 keys x 64 d
      async_ld16(k + base + (size_t)(kb + key) * kDM + c8,
                 &Ks[buf][key * 64 + c8]);
      int dd = e >> 2, v8o = (e & 3) * 8;          // Vt: 64 d x 32 keys
      async_ld16(vt + vbase + (size_t)dd * kT + kb + v8o,
                 &Vt[buf][dd * 32 + v8o]);
    }
#else
    h8 tk[2], tv[2];
#pragma unroll
    for (int i = 0; i < 2; ++i) {
      int e = tid + i * 128;
      int key = e >> 3, c8 = (e & 7) * 8;
      tk[i] = *(const h8*)(k + base + (size_t)(kb + key) * kDM + c8);
      int dd = e >> 2, v8o = (e & 3) * 8;
      tv[i] = *(const h8*)(vt + vbase + (size_t)dd * kT + kb + v8o);
    }
#pragma unroll
    for (int i = 0; i < 2; ++i) {
      int e = tid + i * 128;
      int key = e >> 3, c8 = (e & 7) * 8;
      *(h8*)(&Ks[buf][key * 64 + c8]) = tk[i];
      int dd = e >> 2, v8o = (e & 3) * 8;
      *(h8*)(&Vt[buf][dd * 32 + v8o]) = tv[i];
    }
#endif
  };

  // Q fragments (d-chunks 0-31, 32-63)
  v16h qa[2];
  {
    int row = qrow_w + (lane & 15);
    const _Float16* qp = q + base + (size_t)row * kDM;
#pragma unroll
    for (int c = 0; c < 2; ++c) {
      int kofs = c * 32 + g * 8;
      qa[c] = cat16(*(const h8*)(qp + kofs), *(const h8*)(qp + kofs + 16));
    }
  }

  float mrow[8], lrow[8];
#pragma unroll
  for (int r = 0; r < 8; ++r) { mrow[r] = -1e30f; lrow[r] = 0.f; }
  v8f acc[4] = {v8f{}, v8f{}, v8f{}, v8f{}};

  const int nkb = kT / 32;
  stageKV(0, 0);

  for (int it = 0; it < nkb; ++it) {
    const int buf = it & 1;
    const int kb = it * 32;
    async_wait();
    __syncthreads();
    if (it + 1 < nkb) stageKV(kb + 32, buf ^ 1);

    // scores: two 16x16 fragments over 32 keys, K-dim 64 = 2 chunks
    const int kbv = g * 8;
    const int key0l = lane & 15, key1l = key0l + 16;
    v16h sb[4];
#pragma unroll
    for (int c = 0; c < 2; ++c) {
      sb[c * 2 + 0] = cat16(*(const h8*)&Ks[buf][key0l * 64 + c * 32 + kbv],
                            *(const h8*)&Ks[buf][key0l * 64 + c * 32 + kbv + 16]);
      sb[c * 2 + 1] = cat16(*(const h8*)&Ks[buf][key1l * 64 + c * 32 + kbv],
                            *(const h8*)&Ks[buf][key1l * 64 + c * 32 + kbv + 16]);
    }
    v8f s0 = v8f{}, s1 = v8f{};
    s0 = wmma16(qa[0], sb[0], s0);
    s1 = wmma16(qa[0], sb[1], s1);
    s0 = wmma16(qa[1], sb[2], s0);
    s1 = wmma16(qa[1], sb[3], s1);

    // scale + periodic bias; per-row chunk max (16-lane half reduction)
    int key0 = kb + key0l, key1 = key0 + 16;
    float cmax[8];
#pragma unroll
    for (int r = 0; r < 8; ++r) {
      int trow = qrow_w + r + 8 * g;
      float v0 = s0[r] * 0.125f + bias_s * floorf(fabsf((float)(trow - key0)) * (1.f / 30.f));
      float v1 = s1[r] * 0.125f + bias_s * floorf(fabsf((float)(trow - key1)) * (1.f / 30.f));
      s0[r] = v0; s1[r] = v1;
      float mx = fmaxf(v0, v1);
#pragma unroll
      for (int off = 8; off >= 1; off >>= 1)
        mx = fmaxf(mx, __shfl_xor(mx, off, 32));
      cmax[r] = mx;
    }

    // online softmax; P -> per-wave LDS (same-wave DS round-trip, no barrier)
#pragma unroll
    for (int r = 0; r < 8; ++r) {
      float mnew = fmaxf(mrow[r], cmax[r]);
      float ps = __expf(mrow[r] - mnew);
      mrow[r] = mnew;
      float p0 = __expf(s0[r] - mnew);
      float p1 = __expf(s1[r] - mnew);
      float ssum = p0 + p1;
#pragma unroll
      for (int off = 8; off >= 1; off >>= 1) ssum += __shfl_xor(ssum, off, 32);
      lrow[r] = lrow[r] * ps + ssum;
      int prow = r + 8 * g;
      Pw[wid][prow * 32 + key0l]      = (_Float16)p0;
      Pw[wid][prow * 32 + key0l + 16] = (_Float16)p1;
#pragma unroll
      for (int n = 0; n < 4; ++n) acc[n][r] *= ps;
    }

    // PV: A = P[16x32], B = V[32x64]
    v16h pf;
    {
      int prow = lane & 15, kofs = g * 8;
      pf = cat16(*(const h8*)&Pw[wid][prow * 32 + kofs],
                 *(const h8*)&Pw[wid][prow * 32 + kofs + 16]);
    }
    v16h vf[4];
#pragma unroll
    for (int n = 0; n < 4; ++n) {
      int col = n * 16 + (lane & 15);
      vf[n] = cat16(*(const h8*)&Vt[buf][col * 32 + kbv],
                    *(const h8*)&Vt[buf][col * 32 + kbv + 16]);
    }
#pragma unroll
    for (int n = 0; n < 4; ++n) acc[n] = wmma16(pf, vf[n], acc[n]);
  }

  // normalize, emit f16 row-major (feeds Wo GEMM)
#pragma unroll
  for (int n = 0; n < 4; ++n) {
    int col = n * 16 + (lane & 15);
#pragma unroll
    for (int r = 0; r < 8; ++r) {
      int row = qrow_w + r + 8 * g;
      o[base + (size_t)row * kDM + col] = (_Float16)(acc[n][r] / lrow[r]);
    }
  }
}

// ---------------------------------------------------------------------------
// Output: motion blend + contact sigmoid (dec fp32 [BT,139])
// ---------------------------------------------------------------------------
__global__ void k_final(const float* __restrict__ dec,
                        const float* __restrict__ motion,
                        const float* __restrict__ mask,
                        float* __restrict__ out) {
  int i = blockIdx.x * blockDim.x + threadIdx.x;
  if (i >= kBT * kDOUT) return;
  int row = i / kDOUT, c = i % kDOUT;
  float v = dec[(size_t)row * kDOUT + c];
  if (c < 135) {
    float mk = mask[row];
    out[(size_t)row * 135 + c] = motion[(size_t)row * 135 + c] * mk + v * (1.f - mk);
  } else {
    out[(size_t)kBT * 135 + (size_t)row * 4 + (c - 135)] = 1.f / (1.f + __expf(-v));
  }
}

// ---------------------------------------------------------------------------
// Host orchestration
// ---------------------------------------------------------------------------
extern "C" void kernel_launch(void* const* d_in, const int* in_sizes, int n_in,
                              void* d_out, int out_size, void* d_ws, size_t ws_size,
                              hipStream_t stream) {
  (void)in_sizes; (void)n_in; (void)out_size; (void)ws_size;
  const float* motion = (const float*)d_in[0];
  const float* traj   = (const float*)d_in[1];
  const float* mask   = (const float*)d_in[2];
  const float* encW1  = (const float*)d_in[3];
  const float* encb1  = (const float*)d_in[4];
  const float* enca1  = (const float*)d_in[5];
  const float* encW2  = (const float*)d_in[6];
  const float* encb2  = (const float*)d_in[7];
  const float* enca2  = (const float*)d_in[8];
  const float* aln_g  = (const float*)d_in[9];
  const float* aln_b  = (const float*)d_in[10];
  const float* Wq_all = (const float*)d_in[11];
  const float* bq_all = (const float*)d_in[12];
  const float* Wk_all = (const float*)d_in[13];
  const float* bk_all = (const float*)d_in[14];
  const float* Wv_all = (const float*)d_in[15];
  const float* bv_all = (const float*)d_in[16];
  const float* Wo_all = (const float*)d_in[17];
  const float* bo_all = (const float*)d_in[18];
  const float* fln_g  = (const float*)d_in[19];
  const float* fln_b  = (const float*)d_in[20];
  const float* fW1    = (const float*)d_in[21];
  const float* fb1    = (const float*)d_in[22];
  const float* fW2    = (const float*)d_in[23];
  const float* fb2    = (const float*)d_in[24];
  const float* fing   = (const float*)d_in[25];
  const float* finb   = (const float*)d_in[26];
  const float* decW1  = (const float*)d_in[27];
  const float* decb1  = (const float*)d_in[28];
  const float* deca1  = (const float*)d_in[29];
  const float* decW2  = (const float*)d_in[30];
  const float* decb2  = (const float*)d_in[31];

  // workspace: fp32 X | fp32 scr | f16 weights | f16 activations
  float* X   = (float*)d_ws;
  float* scr = X + (size_t)kBT * kDM;
  _Float16* hp = (_Float16*)(scr + (size_t)kBT * kDM);
  _Float16* wt_e1 = hp;                        hp += (size_t)kDM * kKIN;
  _Float16* wt_e2 = hp;                        hp += (size_t)kDM * kDM;
  _Float16* wt_q  = hp;                        hp += (size_t)kNL * kDM * kDM;
  _Float16* wt_k  = hp;                        hp += (size_t)kNL * kDM * kDM;
  _Float16* wt_v  = hp;                        hp += (size_t)kNL * kDM * kDM;
  _Float16* wt_o  = hp;                        hp += (size_t)kNL * kDM * kDM;
  _Float16* wt_f1 = hp;                        hp += (size_t)kNL * kDFF * kDM;
  _Float16* wt_f2 = hp;                        hp += (size_t)kNL * kDM * kDFF;
  _Float16* wt_d1 = hp;                        hp += (size_t)kDM * kDM;
  _Float16* wt_d2 = hp;                        hp += (size_t)kDOUT * kDM;
  _Float16* x0h   = hp;                        hp += (size_t)kBT * kKIN;
  _Float16* xih   = hp;                        hp += (size_t)kBT * kDM;
  _Float16* aoh   = hp;                        hp += (size_t)kBT * kDM;
  _Float16* bigh  = hp;                        // kBT * kDFF
  _Float16* qh  = bigh;
  _Float16* kh  = bigh + (size_t)kBT * kDM;
  _Float16* vth = bigh + 2 * (size_t)kBT * kDM;   // [b,h,d,t]

  dim3 blk(128);
  auto cvtw = [&](const float* W, _Float16* Wt, int Kb, int N, int Kpad, int L) {
    long tot = (long)L * N * Kpad;
    k_cvt_w<<<(unsigned)((tot + 255) / 256), 256, 0, stream>>>(W, Wt, Kb, N, Kpad, tot);
  };
  auto gemm = [&](const _Float16* A, const _Float16* Bt, const float* bias,
                  const float* res, float* C, _Float16* Ch,
                  int N, int K, int mode, int act, const float* alphap) {
    dim3 grid(kBT / 64, (N + 63) / 64);
    k_gemm<<<grid, blk, 0, stream>>>(A, Bt, bias, res, C, Ch, N, K, mode, act, alphap);
  };

  // weight conversion (transposed f16, K zero-padded)
  cvtw(encW1, wt_e1, kDIN, kDM, kKIN, 1);
  cvtw(encW2, wt_e2, kDM, kDM, kDM, 1);
  cvtw(Wq_all, wt_q, kDM, kDM, kDM, kNL);
  cvtw(Wk_all, wt_k, kDM, kDM, kDM, kNL);
  cvtw(Wv_all, wt_v, kDM, kDM, kDM, kNL);
  cvtw(Wo_all, wt_o, kDM, kDM, kDM, kNL);
  cvtw(fW1, wt_f1, kDM, kDFF, kDM, kNL);
  cvtw(fW2, wt_f2, kDFF, kDM, kDFF, kNL);
  cvtw(decW1, wt_d1, kDM, kDM, kDM, 1);
  cvtw(decW2, wt_d2, kDM, kDOUT, kDM, 1);

  // encoder
  k_build<<<(kBT * kKIN + 255) / 256, 256, 0, stream>>>(motion, traj, mask, x0h);
  gemm(x0h, wt_e1, encb1, nullptr, nullptr, xih, kDM, kKIN, 1, 1, enca1);
  gemm(xih, wt_e2, encb2, nullptr, X, nullptr, kDM, kDM, 0, 1, enca2);
  k_pe<<<(kBT * kDM + 255) / 256, 256, 0, stream>>>(X);

  // layers
  for (int l = 0; l < kNL; ++l) {
    const size_t wofs = (size_t)l * kDM * kDM;
    const size_t bofs = (size_t)l * kDM;
    k_ln<<<kBT / 4, blk, 0, stream>>>(X, aln_g + bofs, aln_b + bofs, xih);
    gemm(xih, wt_q + wofs, bq_all + bofs, nullptr, nullptr, qh,  kDM, kDM, 1, 0, nullptr);
    gemm(xih, wt_k + wofs, bk_all + bofs, nullptr, nullptr, kh,  kDM, kDM, 1, 0, nullptr);
    gemm(xih, wt_v + wofs, bv_all + bofs, nullptr, nullptr, vth, kDM, kDM, 2, 0, nullptr);
    dim3 ga(kT / 64, kB * kNH);
    k_attn<<<ga, blk, 0, stream>>>(qh, kh, vth, aoh);
    gemm(aoh, wt_o + wofs, bo_all + bofs, X, X, nullptr, kDM, kDM, 0, 0, nullptr);
    k_ln<<<kBT / 4, blk, 0, stream>>>(X, fln_g + bofs, fln_b + bofs, xih);
    gemm(xih,  wt_f1 + (size_t)l * kDFF * kDM, fb1 + (size_t)l * kDFF,
         nullptr, nullptr, bigh, kDFF, kDM, 1, 2, nullptr);
    gemm(bigh, wt_f2 + (size_t)l * kDM * kDFF, fb2 + bofs,
         X, X, nullptr, kDM, kDFF, 0, 0, nullptr);
  }

  // decoder
  k_ln<<<kBT / 4, blk, 0, stream>>>(X, fing, finb, xih);
  gemm(xih, wt_d1, decb1, nullptr, nullptr, aoh, kDM, kDM, 1, 1, deca1);
  gemm(aoh, wt_d2, decb2, nullptr, scr, nullptr, kDOUT, kDM, 0, 0, nullptr);
  k_final<<<(kBT * kDOUT + 255) / 256, 256, 0, stream>>>(scr, motion, mask, (float*)d_out);
}
